// GSAT_37598143709469
// MI455X (gfx1250) — matrix operand order repaired and checked by reading, not compiled, not run on hardware
//
#include <hip/hip_runtime.h>
#include <hip/hip_bf16.h>
#include <stdint.h>

// ---------------------------------------------------------------------------
// Problem constants (from reference)
// ---------------------------------------------------------------------------
#define GB   8       // batch
#define GN   1024    // nodes
#define GF   128     // F_IN
#define GH   4       // heads
#define GDH  64      // D_HID
#define GDO  64      // D_OUT
#define EPSF 1.1920929e-07f
#define SLOPE 0.01f

typedef __attribute__((ext_vector_type(16))) __bf16 v16bf;
typedef __attribute__((ext_vector_type(8)))  float  v8f;

__device__ __forceinline__ uint16_t f32_to_bf16(float f) {
    uint32_t u = __builtin_bit_cast(uint32_t, f);
    uint32_t r = u + 0x7FFFu + ((u >> 16) & 1u);   // round-to-nearest-even
    return (uint16_t)(r >> 16);
}

__device__ __forceinline__ uint32_t pack_bf16x2(float a, float b) {
    return (uint32_t)f32_to_bf16(a) | ((uint32_t)f32_to_bf16(b) << 16);
}

__device__ __forceinline__ float hash_u01(uint32_t x) {
    x ^= x >> 17; x *= 0xED5AD4BBu;
    x ^= x >> 11; x *= 0xAC4C1B51u;
    x ^= x >> 15; x *= 0x31848BABu;
    x ^= x >> 14;
    return ((float)(x >> 8) + 0.5f) * (1.0f / 16777216.0f);
}

// ---------------------------------------------------------------------------
// CDNA5 async memory->LDS copy (GLOBAL_LOAD_ASYNC_TO_LDS_B128, ASYNCcnt).
// LDS address operand = low 32 bits of the generic shared pointer
// (ISA 10.2: LDS aperture address = addr[31:0]).
// ---------------------------------------------------------------------------
__device__ __forceinline__ void async_copy_b128(void* lds_dst, const void* gsrc) {
    asm volatile("global_load_async_to_lds_b128 %0, %1, off"
                 :: "v"((uint32_t)(uintptr_t)lds_dst), "v"(gsrc)
                 : "memory");
}
__device__ __forceinline__ void wait_async0() {
    asm volatile("s_wait_asynccnt 0x0" ::: "memory");
}

// ---------------------------------------------------------------------------
// f32 -> bf16 elementwise cast (row-major copy)
// ---------------------------------------------------------------------------
__global__ void cvt_bf16_kernel(const float* __restrict__ in,
                                uint16_t* __restrict__ out, long long n) {
    long long i = (long long)blockIdx.x * blockDim.x + threadIdx.x;
    if (i < n) out[i] = f32_to_bf16(in[i]);
}

// f32[R][C] -> bf16 transposed out[C][R]  (for GEMM B operands)
__global__ void cvt_bf16_t_kernel(const float* __restrict__ in,
                                  uint16_t* __restrict__ out, int R, int C) {
    int i = blockIdx.x * blockDim.x + threadIdx.x;
    if (i < R * C) {
        int r = i / C, c = i % C;
        out[c * R + r] = f32_to_bf16(in[i]);
    }
}

// ---------------------------------------------------------------------------
// Tiled WMMA bf16 GEMM: C = act(A @ B).
//   A : [M,K]  bf16 row-major
//   Bt: [Nc,K] bf16 (B pre-transposed; row n holds column n of B, K contig)
// Block = 128 threads = 4 waves.  Block tile 64(M) x 64(N); wave w owns rows
// 16w..16w+15 and all four 16-col tiles -> 4 WMMAs per K-step per wave with
// the A fragment reused.  Tiles staged with DOUBLE-BUFFERED async b128 LDS
// DMA: the copy for K-step i+1 runs under the WMMAs of K-step i; one barrier
// per K-step.  All B fragments are loaded into distinct register groups
// before the WMMA burst so ds_load waits stagger instead of serializing.
// Requires M%64==0, Nc%64==0, K%32==0 (all GEMMs here comply).
// Outputs (compile-time): f32 row-major, bf16 row-major, bf16 transposed
// ([Nc][M], for consumption as a later Bt operand).
// ---------------------------------------------------------------------------
#define TM 64
#define TN 64
#define TK 32

struct Stage {
    uint16_t a[TM * TK];     // row-major A tile   (4 KB)
    uint16_t b[TN * TK];     // Bt tile sB[n][k]   (4 KB)
};
struct GemmSmem {
    union {
        Stage st[2];                 // double-buffered stages (16 KB)
        float out[TM * TN];          // f32 result tile        (16 KB)
    };
};

__device__ __forceinline__ void stage_tiles(Stage* s,
                                            const uint16_t* __restrict__ A,
                                            const uint16_t* __restrict__ Bt,
                                            int m0, int n0, int K, int ldbt,
                                            int k0, int tid)
{
    #pragma unroll
    for (int i = 0; i < 2; ++i) {
        int ch = tid * 2 + i;            // 0..255
        int r  = ch >> 2;                // row 0..63
        int c4 = ch & 3;                 // 16B chunk within row
        async_copy_b128(((uint4*)s->a) + ch,
                        A + (long long)(m0 + r) * K + k0 + c4 * 8);
        async_copy_b128(((uint4*)s->b) + ch,
                        Bt + (long long)(n0 + r) * ldbt + k0 + c4 * 8);
    }
}

template <bool LEAKY, bool WF32, bool WBF16, bool TBF16>
__global__ __launch_bounds__(128)
void wmma_gemm_bf16(const uint16_t* __restrict__ A, const uint16_t* __restrict__ Bt,
                    float* __restrict__ Cf, uint16_t* __restrict__ Cbf,
                    int M, int Nc, int K, int ldbt,
                    long long strideA, long long strideB, long long strideC)
{
    __shared__ GemmSmem smem;

    const int batch = blockIdx.z;
    A  += (long long)batch * strideA;
    Bt += (long long)batch * strideB;
    const long long coff = (long long)batch * strideC;

    const int tid  = threadIdx.x;
    const int wave = tid >> 5;
    const int lane = tid & 31;
    const int half = lane >> 4;    // 0: lanes 0-15, 1: lanes 16-31
    const int lr   = lane & 15;

    const int m0 = blockIdx.x * TM;
    const int n0 = blockIdx.y * TN;

    v8f acc[4] = {};

    const int nIter = K / TK;
    // prologue: DMA first tile into buffer 0
    stage_tiles(&smem.st[0], A, Bt, m0, n0, K, ldbt, 0, tid);

    for (int it = 0; it < nIter; ++it) {
        const int k0 = it * TK;
        // copies targeting st[it&1] were issued one iteration ago and have
        // been running under the previous iteration's WMMAs
        wait_async0();
        __syncthreads();

        // kick off DMA for the next K-step into the other buffer; it runs
        // under this iteration's fragment loads + WMMAs
        if (it + 1 < nIter) {
            stage_tiles(&smem.st[(it + 1) & 1], A, Bt, m0, n0, K, ldbt,
                        k0 + TK, tid);
            if (it + 2 < nIter) {   // warm L2 for the tile after next
                int ch = tid * 2;
                int r  = ch >> 2, c4 = ch & 3;
                __builtin_prefetch(A + (long long)(m0 + r) * K + (k0 + 2 * TK) + c4 * 8, 0, 1);
                __builtin_prefetch(Bt + (long long)(n0 + r) * ldbt + (k0 + 2 * TK) + c4 * 8, 0, 1);
            }
        }

        const Stage* cs = &smem.st[it & 1];
        // --- fragments per documented 16-bit WMMA VGPR layouts -------------
        // A (16x32): lane(row=lr, half) holds K in {8h..8h+7} U {16+8h..+7}
        union FragU { v16bf v; uint4 u[2]; };
        FragU afrag;
        {
            const uint16_t* ab = cs->a + (wave * 16 + lr) * TK;
            afrag.u[0] = *(const uint4*)(ab + 8 * half);
            afrag.u[1] = *(const uint4*)(ab + 16 + 8 * half);
        }
        // B (32x16): lane(col=lr, half) holds K in {8h..8h+7} U {16+8h..+7}
        // Load ALL four B fragments first (distinct register groups) so the
        // ds_load waits stagger and the 4 WMMAs issue back-to-back.
        FragU bfrag[4];
        #pragma unroll
        for (int t = 0; t < 4; ++t) {
            const uint16_t* bb = cs->b + (t * 16 + lr) * TK;
            bfrag[t].u[0] = *(const uint4*)(bb + 8 * half);
            bfrag[t].u[1] = *(const uint4*)(bb + 16 + 8 * half);
        }
        #pragma unroll
        for (int t = 0; t < 4; ++t)
            acc[t] = __builtin_amdgcn_wmma_f32_16x16x32_bf16(
                         false, afrag.v, false, bfrag[t].v, (short)0, acc[t],
                         false, false);
    }
    __syncthreads();   // stage buffers alias smem.out below

    // --- epilogue: stage full 64x64 f32 tile in LDS ------------------------
    // acc[t] VGPR j holds D[M = wave*16 + 8*half + j][N = t*16 + lr]
    #pragma unroll
    for (int t = 0; t < 4; ++t)
        #pragma unroll
        for (int j = 0; j < 8; ++j)
            smem.out[(wave * 16 + 8 * half + j) * TN + t * 16 + lr] = acc[t][j];
    __syncthreads();

    const int hh = tid & 1;           // 32-element half

    if (WF32 || (WBF16 && !TBF16)) {
        const int r = tid >> 1;       // row 0..63
        const float* srow = smem.out + r * TN + hh * 32;
        const long long gbase = coff + (long long)(m0 + r) * Nc + n0 + hh * 32;
        float v[32];
        #pragma unroll
        for (int q = 0; q < 32; ++q) {
            float x = srow[q];
            v[q] = LEAKY ? ((x > 0.0f) ? x : SLOPE * x) : x;
        }
        if (WF32) {
            #pragma unroll
            for (int q = 0; q < 8; ++q)
                *(float4*)(Cf + gbase + q * 4) =
                    make_float4(v[q*4], v[q*4+1], v[q*4+2], v[q*4+3]);
        }
        if (WBF16 && !TBF16) {
            #pragma unroll
            for (int q = 0; q < 4; ++q) {
                uint4 p;
                p.x = pack_bf16x2(v[q*8+0], v[q*8+1]);
                p.y = pack_bf16x2(v[q*8+2], v[q*8+3]);
                p.z = pack_bf16x2(v[q*8+4], v[q*8+5]);
                p.w = pack_bf16x2(v[q*8+6], v[q*8+7]);
                *(uint4*)(Cbf + gbase + q * 8) = p;
            }
        }
    }
    if (WBF16 && TBF16) {
        // transposed bf16 output Ct[Nc][M]; column n of the tile -> row of Ct
        const int n = tid >> 1;
        float v[32];
        #pragma unroll
        for (int q = 0; q < 32; ++q) {
            float x = smem.out[(hh * 32 + q) * TN + n];
            v[q] = LEAKY ? ((x > 0.0f) ? x : SLOPE * x) : x;
        }
        const long long tbase = (long long)(n0 + n) * M + m0 + hh * 32;
        #pragma unroll
        for (int q = 0; q < 4; ++q) {
            uint4 p;
            p.x = pack_bf16x2(v[q*8+0], v[q*8+1]);
            p.y = pack_bf16x2(v[q*8+2], v[q*8+3]);
            p.z = pack_bf16x2(v[q*8+4], v[q*8+5]);
            p.w = pack_bf16x2(v[q*8+6], v[q*8+7]);
            *(uint4*)(Cbf + tbase + q * 8) = p;
        }
    }
}

// ---------------------------------------------------------------------------
// hsrc[b,n,k] = sum_d h[b,n,k,d]*a_src[k,d];  hdst likewise.
// ---------------------------------------------------------------------------
__global__ void head_proj_kernel(const float* __restrict__ h,
                                 const float* __restrict__ a_src,
                                 const float* __restrict__ a_dst,
                                 float* __restrict__ hsrc,
                                 float* __restrict__ hdst,
                                 int totalBN, int D)
{
    int i = blockIdx.x * blockDim.x + threadIdx.x;   // over B*N*H
    if (i >= totalBN * GH) return;
    int bn = i / GH, k = i % GH;
    const float* row = h + (long long)bn * GH * D + (long long)k * D;
    const float* as = a_src + k * D;
    const float* ad = a_dst + k * D;
    float s = 0.f, d = 0.f;
    for (int j = 0; j < D; ++j) { float v = row[j]; s += v * as[j]; d += v * ad[j]; }
    hsrc[i] = s;
    hdst[i] = d;
}

// ---------------------------------------------------------------------------
// Fused gumbel-softmax attention weights, head-collapsed & transposed:
//   At[b][n][m] = sum_k softmax_n( (hsrc[b,m,k]+hdst[b,n,k])*adj[m,n] + g )
// One 256-thread block per (b,m); scores live in registers; two LDS tree
// reductions (per-head max, per-head sum).
// ---------------------------------------------------------------------------
__global__ __launch_bounds__(256)
void attn_weights_kernel(const float* __restrict__ hsrc,
                         const float* __restrict__ hdst,
                         const float* __restrict__ adj,
                         uint16_t* __restrict__ At,
                         uint32_t seed)
{
    __shared__ float red[256 * GH];

    const int b = blockIdx.x / GN;
    const int m = blockIdx.x % GN;
    const int tid = threadIdx.x;
    const int NPT = GN / 256;   // 4 columns per thread

    float hs[GH];
    #pragma unroll
    for (int k = 0; k < GH; ++k)
        hs[k] = hsrc[((long long)b * GN + m) * GH + k];

    float sc[NPT][GH];
    float lmax[GH];
    #pragma unroll
    for (int k = 0; k < GH; ++k) lmax[k] = -1e30f;

    for (int i = 0; i < NPT; ++i) {
        int n = tid + 256 * i;
        float a = adj[(long long)m * GN + n];
        #pragma unroll
        for (int k = 0; k < GH; ++k) {
            float hd = hdst[((long long)b * GN + n) * GH + k];
            uint32_t idx = (uint32_t)((((b * GN + m) * GN + n) * GH) + k);
            float u = hash_u01(idx + seed * 0x9E3779B9u);
            float g = -logf(-logf(u + EPSF) + EPSF);
            float s = (hs[k] + hd) * a + g;       // TAU == 1
            sc[i][k] = s;
            lmax[k] = fmaxf(lmax[k], s);
        }
    }

    // block-reduce max per head
    #pragma unroll
    for (int k = 0; k < GH; ++k) red[tid * GH + k] = lmax[k];
    __syncthreads();
    for (int s2 = 128; s2 > 0; s2 >>= 1) {
        if (tid < s2)
            #pragma unroll
            for (int k = 0; k < GH; ++k)
                red[tid * GH + k] = fmaxf(red[tid * GH + k], red[(tid + s2) * GH + k]);
        __syncthreads();
    }
    float mk[GH];
    #pragma unroll
    for (int k = 0; k < GH; ++k) mk[k] = red[k];
    __syncthreads();

    // exp + block-reduce sum per head
    float lsum[GH] = {0.f, 0.f, 0.f, 0.f};
    for (int i = 0; i < NPT; ++i)
        #pragma unroll
        for (int k = 0; k < GH; ++k) {
            float e = __expf(sc[i][k] - mk[k]);
            sc[i][k] = e;
            lsum[k] += e;
        }
    #pragma unroll
    for (int k = 0; k < GH; ++k) red[tid * GH + k] = lsum[k];
    __syncthreads();
    for (int s2 = 128; s2 > 0; s2 >>= 1) {
        if (tid < s2)
            #pragma unroll
            for (int k = 0; k < GH; ++k)
                red[tid * GH + k] += red[(tid + s2) * GH + k];
        __syncthreads();
    }
    float inv[GH];
    #pragma unroll
    for (int k = 0; k < GH; ++k) inv[k] = 1.0f / red[k];

    // collapse heads, write transposed (row n, col m) for the next GEMM
    for (int i = 0; i < NPT; ++i) {
        int n = tid + 256 * i;
        float a = 0.f;
        #pragma unroll
        for (int k = 0; k < GH; ++k) a += sc[i][k] * inv[k];
        At[((long long)b * GN + n) * GN + m] = f32_to_bf16(a);
    }
}

// ---------------------------------------------------------------------------
// Host-side orchestration
// ---------------------------------------------------------------------------
template <bool LEAKY, bool WF32, bool WBF16, bool TBF16>
static inline void launch_gemm(const uint16_t* A, const uint16_t* Bt,
                               float* Cf, uint16_t* Cbf,
                               int M, int Nc, int K, int ldbt,
                               long long sA, long long sB, long long sC,
                               int batches, hipStream_t stream)
{
    dim3 grid(M / TM, Nc / TN, batches);
    wmma_gemm_bf16<LEAKY, WF32, WBF16, TBF16>
        <<<grid, 128, 0, stream>>>(A, Bt, Cf, Cbf, M, Nc, K, ldbt, sA, sB, sC);
}

extern "C" void kernel_launch(void* const* d_in, const int* in_sizes, int n_in,
                              void* d_out, int out_size, void* d_ws, size_t ws_size,
                              hipStream_t stream)
{
    const float* x      = (const float*)d_in[0];
    const float* adj    = (const float*)d_in[1];
    const float* W1     = (const float*)d_in[2];
    const float* a_src1 = (const float*)d_in[3];
    const float* a_dst1 = (const float*)d_in[4];
    const float* O1     = (const float*)d_in[5];
    const float* W2     = (const float*)d_in[6];
    const float* a_src2 = (const float*)d_in[7];
    const float* a_dst2 = (const float*)d_in[8];
    const float* O2     = (const float*)d_in[9];
    float* out = (float*)d_out;

    const long long BN = (long long)GB * GN;   // 8192
    const int C1 = GH * GDH;                   // 256
    const int C2 = GH * GDO;                   // 256

    // workspace carve-up (256B aligned)
    uint8_t* w = (uint8_t*)d_ws;
    auto carve = [&](size_t bytes) -> void* {
        void* p = (void*)w;
        w += (bytes + 255) & ~(size_t)255;
        return p;
    };
    uint16_t* xbf  = (uint16_t*)carve(BN * GF * 2);
    uint16_t* W1t  = (uint16_t*)carve((size_t)GF * C1 * 2);      // [256][128]
    uint16_t* O1t  = (uint16_t*)carve((size_t)C1 * GDH * 2);     // [64][256]
    uint16_t* W2t  = (uint16_t*)carve((size_t)GDH * C2 * 2);     // [256][64]
    uint16_t* O2t  = (uint16_t*)carve((size_t)C2 * GDO * 2);     // [64][256]
    float*    h1f  = (float*)   carve(BN * C1 * 4);
    uint16_t* h1t  = (uint16_t*)carve(BN * C1 * 2);              // [256][8192]
    float*    hsrc = (float*)   carve(BN * GH * 4);
    float*    hdst = (float*)   carve(BN * GH * 4);
    uint16_t* At   = (uint16_t*)carve((size_t)GB * GN * GN * 2); // 16 MB, reused
    uint16_t* hpre = (uint16_t*)carve(BN * C1 * 2);              // reused both layers
    uint16_t* hp1  = (uint16_t*)carve(BN * GDH * 2);             // [8192][64] row-major
    float*    h2f  = (float*)   carve(BN * C2 * 4);
    uint16_t* h2t  = (uint16_t*)carve(BN * C2 * 2);              // [256][8192]

    auto cvt = [&](const float* src, uint16_t* dst, long long n) {
        cvt_bf16_kernel<<<(unsigned)((n + 255) / 256), 256, 0, stream>>>(src, dst, n);
    };
    auto cvtT = [&](const float* src, uint16_t* dst, int R, int C) {
        cvt_bf16_t_kernel<<<(unsigned)((R * C + 255) / 256), 256, 0, stream>>>(src, dst, R, C);
    };
    cvt (x,  xbf, BN * GF);
    cvtT(W1, W1t, GF,  C1);
    cvtT(O1, O1t, C1,  GDH);
    cvtT(W2, W2t, GDH, C2);
    cvtT(O2, O2t, C2,  GDO);

    // ---- layer 1 ----
    // h1 = x @ W1   [8192,128]@[128,256] -> f32 + bf16 transposed [256][8192]
    launch_gemm<false, true, true, true>(xbf, W1t, h1f, h1t,
                                         (int)BN, C1, GF, GF, 0, 0, 0, 1, stream);
    head_proj_kernel<<<(unsigned)((BN * GH + 255) / 256), 256, 0, stream>>>(
        h1f, a_src1, a_dst1, hsrc, hdst, (int)BN, GDH);
    attn_weights_kernel<<<(unsigned)(GB * GN), 256, 0, stream>>>(
        hsrc, hdst, adj, At, 0x00C0FFEEu);
    // hp1_pre[b] = At[b] @ h1[b]   batched; Bt = h1t cols b*1024.., ldbt=8192
    launch_gemm<false, false, true, false>(At, h1t, nullptr, hpre,
                                           GN, C1, GN, (int)BN,
                                           (long long)GN * GN, (long long)GN,
                                           (long long)GN * C1, GB, stream);
    // hp1 = leaky_relu(hp1_pre @ O1)  [8192,256]@[256,64] -> bf16 row-major
    launch_gemm<true, false, true, false>(hpre, O1t, nullptr, hp1,
                                          (int)BN, GDH, C1, C1, 0, 0, 0, 1, stream);

    // ---- layer 2 ----
    // h2 = hp1 @ W2  [8192,64]@[64,256] -> f32 + bf16 transposed [256][8192]
    launch_gemm<false, true, true, true>(hp1, W2t, h2f, h2t,
                                         (int)BN, C2, GDH, GDH, 0, 0, 0, 1, stream);
    head_proj_kernel<<<(unsigned)((BN * GH + 255) / 256), 256, 0, stream>>>(
        h2f, a_src2, a_dst2, hsrc, hdst, (int)BN, GDO);
    attn_weights_kernel<<<(unsigned)(GB * GN), 256, 0, stream>>>(
        hsrc, hdst, adj, At, 0xDEADBEEFu);
    launch_gemm<false, false, true, false>(At, h2t, nullptr, hpre,
                                           GN, C2, GN, (int)BN,
                                           (long long)GN * GN, (long long)GN,
                                           (long long)GN * C2, GB, stream);
    // out = leaky_relu(hp2_pre @ O2)  [8192,256]@[256,64] -> f32 d_out
    launch_gemm<true, true, false, false>(hpre, O2t, out, nullptr,
                                          (int)BN, GDO, C2, C2, 0, 0, 0, 1, stream);
}
